// LGEA_20023137534372
// MI455X (gfx1250) — matrix operand order repaired and checked by reading, not compiled, not run on hardware
//
#include <hip/hip_runtime.h>

// ---------------------------------------------------------------------------
// Types for CDNA5 WMMA (wave32, 16x16x32 bf16 -> f32)
// ---------------------------------------------------------------------------
typedef __attribute__((ext_vector_type(16))) __bf16 bf16x16;
typedef __attribute__((ext_vector_type(8)))  float  f32x8;

union AFrag { bf16x16 v; unsigned short u[16]; uint4 o[2]; };

__device__ __forceinline__ unsigned short f2bf(float f) {
  unsigned u = __float_as_uint(f);
  unsigned r = (u >> 16) & 1u;
  u += 0x7fffu + r;                 // round-to-nearest-even
  return (unsigned short)(u >> 16);
}
__device__ __forceinline__ unsigned packbf2(float a, float b) {
  return (unsigned)f2bf(a) | ((unsigned)f2bf(b) << 16);
}

// order-preserving float<->uint mapping for atomicMax on floats
__device__ __forceinline__ unsigned encf(float x) {
  unsigned u = __float_as_uint(x);
  return (u & 0x80000000u) ? ~u : (u | 0x80000000u);
}
__device__ __forceinline__ float decf(unsigned u) {
  return __uint_as_float((u & 0x80000000u) ? (u & 0x7fffffffu) : ~u);
}

__device__ __forceinline__ float block_sum_256(float v, float* red) {
  int t = threadIdx.x;
  red[t] = v; __syncthreads();
#pragma unroll
  for (int s = 128; s > 0; s >>= 1) {
    if (t < s) red[t] += red[t + s];
    __syncthreads();
  }
  float r = red[0];
  __syncthreads();
  return r;
}

// ---------------------------------------------------------------------------
// Generic tiled bf16-WMMA GEMM:  C[M,Nc] = A[M,K] @ B[K,Nc]   (fp32 in/out)
// block = 256 threads (8 wave32s), tile 128x128, K-step 32.
// Waves arranged 2(m) x 4(n); each wave computes 64x32 = 4x2 WMMA tiles.
// Software-pipelined: next K-tile's global loads are issued into registers
// before the current tile's fragment loads + WMMAs (latency hiding).
// A staged row-major bf16 in LDS; B staged COLUMN-major so each lane's
// B fragment (16 K-consecutive values) is two 16B ds_load_b128.
// ---------------------------------------------------------------------------
enum { GF_ATOMIC = 1, GF_ROWSCALE = 2, GF_BIAS = 4, GF_RESID = 8, GF_MIX = 16 };

constexpr int TBM = 128, TBN = 128, TBK = 32;
constexpr int LDA_S = TBK + 8;   // 40 ushorts = 80 B row stride (16B-aligned)
constexpr int LDB_S = TBK + 8;   // Bs is [TBN][LDB_S] column-major tile

__global__ __launch_bounds__(256)
void gemm_wmma(const float* __restrict__ A, const float* __restrict__ B,
               float* __restrict__ C,
               int M, int Nc, int K, int lda, int ldb, int ldc,
               int kc, int flags,
               const float* __restrict__ rowscale,
               const float* __restrict__ bias,
               const float* __restrict__ resid, int ldr,
               const float* __restrict__ alpha_p)
{
  __shared__ __align__(16) unsigned short As[TBM * LDA_S];   // 10240 B
  __shared__ __align__(16) unsigned short Bs[TBN * LDB_S];   // 10240 B

  const int tid  = threadIdx.x;
  const int m0   = blockIdx.y * TBM;
  const int n0   = blockIdx.x * TBN;
  const int kbeg = blockIdx.z * kc;
  const int kend = (K < kbeg + kc) ? K : (kbeg + kc);

  const int wid  = tid >> 5;          // wave32
  const int lane = tid & 31;
  const int wm   = wid >> 2;          // 0..1  (row groups of 64)
  const int wn   = wid & 3;           // 0..3  (col groups of 32)
  const int lr   = lane & 15;         // A-row / B-col / C-col within 16
  const int hi   = lane >> 4;         // lane half
  const int koffA = hi << 4;          // A K split granularity 16
  const int koffB = hi << 4;          // B K split granularity 16

  f32x8 acc[4][2];
#pragma unroll
  for (int a = 0; a < 4; ++a)
#pragma unroll
    for (int b = 0; b < 2; ++b)
#pragma unroll
      for (int v = 0; v < 8; ++v) acc[a][b][v] = 0.f;

  const bool mfull = (m0 + TBM <= M);
  const bool nfull = (n0 + TBN <= Nc);
  const bool al4   = ((lda & 3) == 0);

  // per-thread register tile (double-buffer stage)
  float4 aReg[4];
  float  bReg[16];

  // gather one K-tile (k0) into registers
  auto loadTile = [&](int k0) {
    const bool interior = mfull && nfull && al4 && (k0 + TBK <= kend);
    if (interior) {
#pragma unroll
      for (int i = 0; i < 4; ++i) {
        int idx4 = tid + i * 256;                 // 0..1023 float4 ids
        int r = idx4 >> 3, c4 = (idx4 & 7) << 2; // 8 float4 per 32-wide row
        const float* ap = A + (size_t)(m0 + r) * lda + k0 + c4;
        aReg[i] = *(const float4*)ap;
        __builtin_prefetch(ap + 2 * TBK, 0, 1);  // speculative, 2 tiles ahead
      }
#pragma unroll
      for (int i = 0; i < 2; ++i) {
        int t2 = tid + i * 256;                   // 0..511
        int n = t2 & 127, ko = (t2 >> 7) << 3;    // ko in {0,8,16,24}
        const float* bp = B + (size_t)(k0 + ko) * ldb + n0 + n;
#pragma unroll
        for (int j = 0; j < 8; ++j) bReg[i * 8 + j] = bp[(size_t)j * ldb];
      }
    } else {
      // boundary: guarded scalar gather into the same register layout
#pragma unroll
      for (int i = 0; i < 4; ++i) {
        int idx4 = tid + i * 256;
        int r = idx4 >> 3, c4 = (idx4 & 7) << 2;
        int gm = m0 + r;
        const float* ap = A + (size_t)gm * lda + k0 + c4;
        float4 vv;
        vv.x = (gm < M && k0 + c4 + 0 < kend) ? ap[0] : 0.f;
        vv.y = (gm < M && k0 + c4 + 1 < kend) ? ap[1] : 0.f;
        vv.z = (gm < M && k0 + c4 + 2 < kend) ? ap[2] : 0.f;
        vv.w = (gm < M && k0 + c4 + 3 < kend) ? ap[3] : 0.f;
        aReg[i] = vv;
      }
#pragma unroll
      for (int i = 0; i < 2; ++i) {
        int t2 = tid + i * 256;
        int n = t2 & 127, ko = (t2 >> 7) << 3;
        int gn = n0 + n;
#pragma unroll
        for (int j = 0; j < 8; ++j) {
          int gk = k0 + ko + j;
          bReg[i * 8 + j] = (gk < kend && gn < Nc) ? B[(size_t)gk * ldb + gn] : 0.f;
        }
      }
    }
  };

  // convert + store register tile to LDS
  auto storeTile = [&]() {
#pragma unroll
    for (int i = 0; i < 4; ++i) {
      int idx4 = tid + i * 256;
      int r = idx4 >> 3, c4 = (idx4 & 7) << 2;
      uint2 p;
      p.x = packbf2(aReg[i].x, aReg[i].y);
      p.y = packbf2(aReg[i].z, aReg[i].w);
      *(uint2*)&As[r * LDA_S + c4] = p;
    }
#pragma unroll
    for (int i = 0; i < 2; ++i) {
      int t2 = tid + i * 256;
      int n = t2 & 127, ko = (t2 >> 7) << 3;
      uint4 pk;
      pk.x = packbf2(bReg[i * 8 + 0], bReg[i * 8 + 1]);
      pk.y = packbf2(bReg[i * 8 + 2], bReg[i * 8 + 3]);
      pk.z = packbf2(bReg[i * 8 + 4], bReg[i * 8 + 5]);
      pk.w = packbf2(bReg[i * 8 + 6], bReg[i * 8 + 7]);
      *(uint4*)&Bs[n * LDB_S + ko] = pk;
    }
  };

  loadTile(kbeg);                                 // prologue
  for (int k0 = kbeg; k0 < kend; k0 += TBK) {
    storeTile();
    __syncthreads();
    if (k0 + TBK < kend) loadTile(k0 + TBK);      // overlap with WMMA below

    // ---- fragments: all 16B-aligned LDS vector loads
    AFrag af[4], bfr[2];
#pragma unroll
    for (int tn = 0; tn < 2; ++tn) {
      int base = (wn * 32 + tn * 16 + lr) * LDB_S + koffB;
      bfr[tn].o[0] = *(const uint4*)&Bs[base];
      bfr[tn].o[1] = *(const uint4*)&Bs[base + 8];
    }
#pragma unroll
    for (int tm = 0; tm < 4; ++tm) {
      int base = (wm * 64 + tm * 16 + lr) * LDA_S + koffA;
      af[tm].o[0] = *(const uint4*)&As[base];
      af[tm].o[1] = *(const uint4*)&As[base + 8];
    }
#pragma unroll
    for (int tm = 0; tm < 4; ++tm)
#pragma unroll
      for (int tn = 0; tn < 2; ++tn)
        acc[tm][tn] = __builtin_amdgcn_wmma_f32_16x16x32_bf16(
            false, af[tm].v, false, bfr[tn].v, (short)0, acc[tm][tn],
            false, false);
    __syncthreads();
  }

  // ---- epilogue ----
  float al = 0.f;
  if (flags & GF_MIX) al = *alpha_p;
#pragma unroll
  for (int tm = 0; tm < 4; ++tm)
#pragma unroll
    for (int tn = 0; tn < 2; ++tn)
#pragma unroll
      for (int v = 0; v < 8; ++v) {
        int gm = m0 + wm * 64 + tm * 16 + (hi << 3) + v;
        int gn = n0 + wn * 32 + tn * 16 + lr;
        if ((mfull && nfull) || (gm < M && gn < Nc)) {
          float x = acc[tm][tn][v];
          if (flags & GF_BIAS)     x += bias[gn];
          if (flags & GF_ROWSCALE) x *= rowscale[gm];
          if (flags & GF_RESID)    x += resid[(size_t)gm * ldr + gn];
          if (flags & GF_MIX) {
            x = al * x + (1.f - al) * resid[(size_t)gm * ldr + gn];
            x = fmaxf(x, 0.f);
          }
          if (flags & GF_ATOMIC) atomicAdd(&C[(size_t)gm * ldc + gn], x);
          else                   C[(size_t)gm * ldc + gn] = x;
        }
      }
}

// ---------------------------------------------------------------------------
// Small helper kernels
// ---------------------------------------------------------------------------
__global__ void fill_f32(float* p, float v, long long n) {
  long long i = (long long)blockIdx.x * 256 + threadIdx.x;
  if (i < n) p[i] = v;
}
__global__ void fill_u32(unsigned* p, unsigned v, long long n) {
  long long i = (long long)blockIdx.x * 256 + threadIdx.x;
  if (i < n) p[i] = v;
}

// 1/(rowsum + 1e-5) over K columns; one block (256) per row.
__global__ __launch_bounds__(256)
void rowsum_inv(const float* __restrict__ A, float* __restrict__ out, int K) {
  __shared__ float red[256];
  int n = blockIdx.x;
  float s = 0.f;
  for (int k = threadIdx.x; k < K; k += 256) s += A[(size_t)n * K + k];
  float tot = block_sum_256(s, red);
  if (threadIdx.x == 0) out[n] = 1.f / (tot + 1e-5f);
}

// pack gat_W[l] : [H,D,D] -> [D, H*D]
__global__ void pack_gatW(const float* __restrict__ W, float* __restrict__ out) {
  int idx = blockIdx.x * 256 + threadIdx.x;     // < 256*1024
  int e  = idx & 255;
  int hh = (idx >> 8) & 3;
  int dd = idx >> 10;
  out[(size_t)dd * 1024 + hh * 256 + e] = W[(size_t)hh * 65536 + dd * 256 + e];
}

// es/ed per (node, head); block=128 (4 waves, one head per wave)
__global__ __launch_bounds__(128)
void gat_coef(const float* __restrict__ h, const float* __restrict__ a_s,
              const float* __restrict__ a_d, float* es, float* ed) {
  int n = blockIdx.x, hh = threadIdx.x >> 5, lane = threadIdx.x & 31;
  const float* hp = h + (size_t)n * 1024 + hh * 256;
  float s = 0.f, d = 0.f;
#pragma unroll
  for (int j = 0; j < 8; ++j) {
    float v = hp[lane + 32 * j];
    s += v * a_s[hh * 256 + lane + 32 * j];
    d += v * a_d[hh * 256 + lane + 32 * j];
  }
#pragma unroll
  for (int off = 16; off; off >>= 1) {
    s += __shfl_down(s, off, 32);
    d += __shfl_down(d, off, 32);
  }
  if (lane == 0) { es[n * 4 + hh] = s; ed[n * 4 + hh] = d; }
}

__global__ void edge_max(const float* __restrict__ es, const float* __restrict__ ed,
                         const int* __restrict__ src, const int* __restrict__ dst,
                         unsigned* __restrict__ mx, int E) {
  int idx = blockIdx.x * 256 + threadIdx.x;
  if (idx >= E * 4) return;
  int hh = idx & 3, e = idx >> 2;
  float v = es[src[e] * 4 + hh] + ed[dst[e] * 4 + hh];
  v = v > 0.f ? v : 0.2f * v;
  atomicMax(mx + dst[e] * 4 + hh, encf(v));
}

__global__ void edge_exp(const float* __restrict__ es, const float* __restrict__ ed,
                         const int* __restrict__ src, const int* __restrict__ dst,
                         const unsigned* __restrict__ mx, float* __restrict__ den,
                         float* __restrict__ exb, int E) {
  int idx = blockIdx.x * 256 + threadIdx.x;
  if (idx >= E * 4) return;
  int hh = idx & 3, e = idx >> 2;
  float v = es[src[e] * 4 + hh] + ed[dst[e] * 4 + hh];
  v = v > 0.f ? v : 0.2f * v;
  float ex = __expf(v - decf(mx[dst[e] * 4 + hh]));
  exb[idx] = ex;
  atomicAdd(den + dst[e] * 4 + hh, ex);
}

// one block (256) per edge: out[dst] += attn * h[src]  (4 heads x 256 ch)
__global__ __launch_bounds__(256)
void edge_scatter(const float* __restrict__ h, const float* __restrict__ exb,
                  const float* __restrict__ den, const int* __restrict__ src,
                  const int* __restrict__ dst, float* __restrict__ out) {
  int e = blockIdx.x;
  int hh = threadIdx.x >> 6;
  int q  = (threadIdx.x & 63) * 4;
  int s = src[e], d = dst[e];
  float a = exb[(size_t)e * 4 + hh] / (den[d * 4 + hh] + 1e-16f);
  const float* hp = h + (size_t)s * 1024 + hh * 256 + q;
  float* op       = out + (size_t)d * 1024 + hh * 256 + q;
#pragma unroll
  for (int j = 0; j < 4; ++j) atomicAdd(op + j, a * hp[j]);
}

// elu -> head mean -> row L2-normalize; block(256) per node
__global__ __launch_bounds__(256)
void gat_finalize(const float* __restrict__ go, float* __restrict__ xo) {
  __shared__ float red[256];
  int n = blockIdx.x, d = threadIdx.x;
  float s = 0.f;
#pragma unroll
  for (int hh = 0; hh < 4; ++hh) {
    float v = go[(size_t)n * 1024 + hh * 256 + d];
    s += (v > 0.f) ? v : (__expf(v) - 1.f);
  }
  s *= 0.25f;
  float ss = block_sum_256(s * s, red);
  xo[(size_t)n * 256 + d] = s / fmaxf(sqrtf(ss), 1e-12f);
}

// stack [x0|x1|x2] chunk rows -> xs[(i*3+r)*256 + d]
__global__ void pack3(const float* __restrict__ x0, const float* __restrict__ x1,
                      const float* __restrict__ x2, float* __restrict__ xs,
                      int n0, int nc) {
  int idx = blockIdx.x * 256 + threadIdx.x;
  if (idx >= nc * 3 * 256) return;
  int d = idx & 255, row = idx >> 8;
  int r = row % 3, i = row / 3;
  const float* srcp = (r == 0) ? x0 : (r == 1) ? x1 : x2;
  xs[idx] = srcp[(size_t)(n0 + i) * 256 + d];
}

// tiny R=3 attention per node; block=128 (one head per wave)
__global__ __launch_bounds__(128)
void mha_attn(const float* __restrict__ qb, const float* __restrict__ kb,
              const float* __restrict__ vb, float* __restrict__ ob) {
  int i = blockIdx.x, hh = threadIdx.x >> 5, lane = threadIdx.x & 31;
  const size_t base = (size_t)i * 3 * 1024 + hh * 256;
  float s[3][3];
#pragma unroll
  for (int rq = 0; rq < 3; ++rq)
#pragma unroll
    for (int rk = 0; rk < 3; ++rk) {
      const float* qp = qb + base + rq * 1024;
      const float* kp = kb + base + rk * 1024;
      float p = 0.f;
#pragma unroll
      for (int j = 0; j < 8; ++j) p += qp[lane + 32 * j] * kp[lane + 32 * j];
#pragma unroll
      for (int off = 16; off; off >>= 1) p += __shfl_xor(p, off, 32);
      s[rq][rk] = p * (1.f / 16.f);     // / sqrt(256)
    }
#pragma unroll
  for (int rq = 0; rq < 3; ++rq) {
    float m = fmaxf(s[rq][0], fmaxf(s[rq][1], s[rq][2]));
    float e0 = __expf(s[rq][0] - m), e1 = __expf(s[rq][1] - m), e2 = __expf(s[rq][2] - m);
    float inv = 1.f / (e0 + e1 + e2);
    float a0 = e0 * inv, a1 = e1 * inv, a2 = e2 * inv;
    const float* v0 = vb + base;
    const float* v1 = vb + base + 1024;
    const float* v2 = vb + base + 2048;
    float* op = ob + base + rq * 1024;
#pragma unroll
    for (int j = 0; j < 8; ++j) {
      int d = lane + 32 * j;
      op[d] = a0 * v0[d] + a1 * v1[d] + a2 * v2[d];
    }
  }
}

// post-norm LN over D=256, then mean over R=3; block(256) per node
__global__ __launch_bounds__(256)
void ln_mean(const float* __restrict__ yb, const float* __restrict__ g,
             const float* __restrict__ b, float* __restrict__ out, int n0) {
  __shared__ float red[256];
  int i = blockIdx.x, d = threadIdx.x;
  float acc = 0.f;
  for (int r = 0; r < 3; ++r) {
    float v = yb[((size_t)i * 3 + r) * 256 + d];
    float mu = block_sum_256(v, red) * (1.f / 256.f);
    float c = v - mu;
    float var = block_sum_256(c * c, red) * (1.f / 256.f);
    acc += c * rsqrtf(var + 1e-5f) * g[d] + b[d];
  }
  out[(size_t)(n0 + i) * 256 + d] = acc * (1.f / 3.f);
}

// final concat into d_out: 4 parts of [N,768]
__global__ void assemble(const float* __restrict__ sr_m, const float* __restrict__ tg_m,
                         const float* __restrict__ rel_sr, const float* __restrict__ rel_tg,
                         const float* __restrict__ e2_sr, const float* __restrict__ e2_tg,
                         const float* __restrict__ ent_sr, const float* __restrict__ ent_tg,
                         float* __restrict__ out, int N) {
  long long idx = (long long)blockIdx.x * 256 + threadIdx.x;
  long long tot = (long long)4 * N * 768;
  if (idx >= tot) return;
  int part = (int)(idx / ((long long)N * 768));
  long long rem = idx % ((long long)N * 768);
  int n = (int)(rem / 768), c = (int)(rem % 768);
  float v;
  const float* relp = (part == 0 || part == 2) ? rel_sr : rel_tg;
  if (c >= 256)        v = relp[(size_t)n * 512 + (c - 256)];
  else if (part == 0)  v = sr_m[(size_t)n * 256 + c];
  else if (part == 1)  v = tg_m[(size_t)n * 256 + c];
  else if (part == 2)  v = e2_sr[(size_t)n * 256 + c] + ent_sr[(size_t)n * 256 + c];
  else                 v = e2_tg[(size_t)n * 256 + c] + ent_tg[(size_t)n * 256 + c];
  out[idx] = v;
}

// ---------------------------------------------------------------------------
// Host orchestration
// ---------------------------------------------------------------------------
static inline int cdiv(long long a, long long b) { return (int)((a + b - 1) / b); }

static inline void launch_gemm(const float* A, const float* B, float* C,
                               int M, int Nc, int K, int lda, int ldb, int ldc,
                               int flags, const float* rowscale, const float* bias,
                               const float* resid, int ldr, const float* alpha_p,
                               int kc, int ksplit, hipStream_t s) {
  dim3 grid(cdiv(Nc, TBN), cdiv(M, TBM), ksplit);
  gemm_wmma<<<grid, 256, 0, s>>>(A, B, C, M, Nc, K, lda, ldb, ldc, kc, flags,
                                 rowscale, bias, resid, ldr, alpha_p);
}

extern "C" void kernel_launch(void* const* d_in, const int* in_sizes, int n_in,
                              void* d_out, int out_size, void* d_ws, size_t ws_size,
                              hipStream_t stream) {
  const float* ent_sr = (const float*)d_in[0];
  const float* ent_tg = (const float*)d_in[1];
  const float* rel_sr = (const float*)d_in[2];
  const float* rel_tg = (const float*)d_in[3];
  const float* alpha  = (const float*)d_in[4];
  const float* ums_sr = (const float*)d_in[5];
  const float* vt_sr  = (const float*)d_in[6];
  const float* ums_tg = (const float*)d_in[7];
  const float* vt_tg  = (const float*)d_in[8];
  const float* adj_sr_in  = (const float*)d_in[9];
  const float* adj_sr_out = (const float*)d_in[10];
  const float* adj_tg_in  = (const float*)d_in[11];
  const float* adj_tg_out = (const float*)d_in[12];
  const float* gatW = (const float*)d_in[13];
  const float* ga_s = (const float*)d_in[14];
  const float* ga_d = (const float*)d_in[15];
  const float* wq = (const float*)d_in[16]; const float* bq = (const float*)d_in[17];
  const float* wk = (const float*)d_in[18]; const float* bk = (const float*)d_in[19];
  const float* wv = (const float*)d_in[20]; const float* bv = (const float*)d_in[21];
  const float* fcw = (const float*)d_in[22]; const float* fcb = (const float*)d_in[23];
  const float* lng = (const float*)d_in[24]; const float* lnb = (const float*)d_in[25];
  const int* edges_sr = (const int*)d_in[26];
  const int* edges_tg = (const int*)d_in[27];

  const int N  = in_sizes[0] / 256;
  const int RN = in_sizes[2] / 256;
  const int Q  = in_sizes[5] / N;
  const int E  = in_sizes[26] / 2;
  const int NC = 2048;                       // MHA node chunk

  // ---- workspace arena ----
  float* wbase = (float*)d_ws;
  size_t off = 0;
  auto alloc = [&](size_t n) -> float* {
    float* p = wbase + off;
    off += (n + 255) & ~(size_t)255;
    return p;
  };
  // persistent
  float* relA_sr = alloc((size_t)N * 512);
  float* relA_tg = alloc((size_t)N * 512);
  float* e2_sr   = alloc((size_t)N * 256);
  float* e2_tg   = alloc((size_t)N * 256);
  float* x1_sr = alloc((size_t)N * 256);
  float* x2_sr = alloc((size_t)N * 256);
  float* x1_tg = alloc((size_t)N * 256);
  float* x2_tg = alloc((size_t)N * 256);
  float* mean_sr = alloc((size_t)N * 256);
  float* mean_tg = alloc((size_t)N * 256);
  const size_t mark = off;

  // scratch budget check (largest phase = GAT)
  size_t needC = (size_t)256 * 1024 + 2 * (size_t)N * 1024 + 4 * (size_t)N * 4 + (size_t)E * 4 + 2048;
  size_t needD = 2 * (size_t)3 * NC * 256 + 4 * (size_t)3 * NC * 1024 + 2048;
  size_t need  = mark + (needC > needD ? needC : needD);
  if (ws_size / 4 < need) return;

  // ================= Phase A: relation aggregation =================
  {
    off = mark;
    float* rsinv = alloc(N);
    struct { const float* adj; const float* rel; float* out; } jobs[4] = {
      { adj_sr_in,  rel_sr, relA_sr       },
      { adj_sr_out, rel_sr, relA_sr + 256 },
      { adj_tg_in,  rel_tg, relA_tg       },
      { adj_tg_out, rel_tg, relA_tg + 256 },
    };
    for (int j = 0; j < 4; ++j) {
      rowsum_inv<<<N, 256, 0, stream>>>(jobs[j].adj, rsinv, RN);
      launch_gemm(jobs[j].adj, jobs[j].rel, jobs[j].out, N, 256, RN, RN, 256, 512,
                  GF_ROWSCALE, rsinv, nullptr, nullptr, 0, nullptr, RN, 1, stream);
    }
  }

  // ================= Phase B: SVD propagation ======================
  for (int g = 0; g < 2; ++g) {
    off = mark;
    float* svd_t = alloc((size_t)Q * 256);
    float* e1    = alloc((size_t)N * 256);
    const float* ent = g ? ent_tg : ent_sr;
    const float* vt  = g ? vt_tg  : vt_sr;
    const float* ums = g ? ums_tg : ums_sr;
    float* e2 = g ? e2_tg : e2_sr;

    const float* e_cur = ent;
    const int KC = 1024;                        // multiple of TBK
    const int ksplit = cdiv(N, KC);
    for (int l = 0; l < 2; ++l) {
      fill_f32<<<cdiv((long long)Q * 256, 256), 256, 0, stream>>>(svd_t, 0.f, (long long)Q * 256);
      // t = vt @ e   (split-K, atomic accumulate)
      launch_gemm(vt, e_cur, svd_t, Q, 256, N, N, 256, 256,
                  GF_ATOMIC, nullptr, nullptr, nullptr, 0, nullptr, KC, ksplit, stream);
      // e_next = relu(alpha * (ums @ t) + (1-alpha) * e_cur)
      float* e_next = (l == 0) ? e1 : e2;
      launch_gemm(ums, svd_t, e_next, N, 256, Q, Q, 256, 256,
                  GF_MIX, nullptr, nullptr, e_cur, 256, alpha, Q, 1, stream);
      e_cur = e_next;
    }
  }

  // ================= Phase C: GAT layers ===========================
  for (int g = 0; g < 2; ++g) {
    off = mark;
    float*    Wpack  = alloc((size_t)256 * 1024);
    float*    h      = alloc((size_t)N * 1024);
    float*    gatout = alloc((size_t)N * 1024);
    float*    es     = alloc((size_t)N * 4);
    float*    ed     = alloc((size_t)N * 4);
    unsigned* mx     = (unsigned*)alloc((size_t)N * 4);
    float*    den    = alloc((size_t)N * 4);
    float*    exb    = alloc((size_t)E * 4);

    const float* ent = g ? ent_tg : ent_sr;
    const int* src = (g ? edges_tg : edges_sr);
    const int* dst = src + E;
    float* xouts[2] = { g ? x1_tg : x1_sr, g ? x2_tg : x2_sr };

    const float* x_in = ent;
    for (int l = 0; l < 2; ++l) {
      pack_gatW<<<1024, 256, 0, stream>>>(gatW + (size_t)l * 4 * 65536, Wpack);
      launch_gemm(x_in, Wpack, h, N, 1024, 256, 256, 1024, 1024,
                  0, nullptr, nullptr, nullptr, 0, nullptr, 256, 1, stream);
      gat_coef<<<N, 128, 0, stream>>>(h, ga_s + (size_t)l * 1024, ga_d + (size_t)l * 1024, es, ed);
      fill_u32<<<cdiv((long long)N * 4, 256), 256, 0, stream>>>(mx, 0u, (long long)N * 4);
      fill_f32<<<cdiv((long long)N * 4, 256), 256, 0, stream>>>(den, 0.f, (long long)N * 4);
      edge_max<<<cdiv((long long)E * 4, 256), 256, 0, stream>>>(es, ed, src, dst, mx, E);
      edge_exp<<<cdiv((long long)E * 4, 256), 256, 0, stream>>>(es, ed, src, dst, mx, den, exb, E);
      fill_f32<<<cdiv((long long)N * 1024, 256), 256, 0, stream>>>(gatout, 0.f, (long long)N * 1024);
      edge_scatter<<<E, 256, 0, stream>>>(h, exb, den, src, dst, gatout);
      gat_finalize<<<N, 256, 0, stream>>>(gatout, xouts[l]);
      x_in = xouts[l];
    }
  }

  // ================= Phase D: multi-range MHA ======================
  for (int g = 0; g < 2; ++g) {
    off = mark;
    float* xs = alloc((size_t)3 * NC * 256);
    float* qb = alloc((size_t)3 * NC * 1024);
    float* kb = alloc((size_t)3 * NC * 1024);
    float* vb = alloc((size_t)3 * NC * 1024);
    float* ob = alloc((size_t)3 * NC * 1024);
    float* yb = alloc((size_t)3 * NC * 256);

    const float* x0 = g ? ent_tg : ent_sr;
    const float* x1 = g ? x1_tg : x1_sr;
    const float* x2 = g ? x2_tg : x2_sr;
    float* meanp = g ? mean_tg : mean_sr;

    for (int n0 = 0; n0 < N; n0 += NC) {
      int nc = (N - n0 < NC) ? (N - n0) : NC;
      int rows = 3 * nc;
      pack3<<<cdiv((long long)rows * 256, 256), 256, 0, stream>>>(x0, x1, x2, xs, n0, nc);
      launch_gemm(xs, wq, qb, rows, 1024, 256, 256, 1024, 1024,
                  GF_BIAS, nullptr, bq, nullptr, 0, nullptr, 256, 1, stream);
      launch_gemm(xs, wk, kb, rows, 1024, 256, 256, 1024, 1024,
                  GF_BIAS, nullptr, bk, nullptr, 0, nullptr, 256, 1, stream);
      launch_gemm(xs, wv, vb, rows, 1024, 256, 256, 1024, 1024,
                  GF_BIAS, nullptr, bv, nullptr, 0, nullptr, 256, 1, stream);
      mha_attn<<<nc, 128, 0, stream>>>(qb, kb, vb, ob);
      launch_gemm(ob, fcw, yb, rows, 256, 1024, 1024, 256, 256,
                  GF_BIAS | GF_RESID, nullptr, fcb, xs, 256, nullptr, 1024, 1, stream);
      ln_mean<<<nc, 256, 0, stream>>>(yb, lng, lnb, meanp, n0);
    }
  }

  // ================= Phase E: assemble outputs =====================
  assemble<<<cdiv((long long)4 * N * 768, 256), 256, 0, stream>>>(
      mean_sr, mean_tg, relA_sr, relA_tg, e2_sr, e2_tg, ent_sr, ent_tg,
      (float*)d_out, N);
}
